// BankModulatedConv_15212774162745
// MI455X (gfx1250) — matrix operand order repaired and checked by reading, not compiled, not run on hardware
//
#include <hip/hip_runtime.h>

typedef __bf16 v16bf __attribute__((ext_vector_type(16)));
typedef __bf16 v8bf  __attribute__((ext_vector_type(8)));
typedef __bf16 v4bf  __attribute__((ext_vector_type(4)));
typedef float  v8f   __attribute__((ext_vector_type(8)));

constexpr int kB = 8, kF = 16, kD = 256, kH = 64, kW = 64;
constexpr int kTaps = 9;
constexpr int kKlen = kTaps * kD;   // 2304 K-dim per output channel
constexpr int XS    = 66;           // x slots: x = -1 .. 64
constexpr int LPAD  = 40;           // padded channel extent in LDS (32 used)

// ---------------------------------------------------------------------------
// Kernel 1: softmax bank mix + style modulation + L2 demodulation.
// Output: bf16 Kw[b][o][tap][i]  (row length 2304, tap-outer/channel-inner)
// ---------------------------------------------------------------------------
__global__ __launch_bounds__(256) void build_kernels(
    const float* __restrict__ bank_request,
    const float* __restrict__ style,
    const float* __restrict__ bank_weight,
    __bf16* __restrict__ Kw)
{
  const int o = blockIdx.x;
  const int b = blockIdx.y;
  const int i = threadIdx.x;  // input channel handled by this thread

  // redundant per-thread softmax over the 16 bank logits (cheap)
  float w[kF];
  float mx = -3.4e38f;
  for (int f = 0; f < kF; ++f) { w[f] = bank_request[b * kF + f]; mx = fmaxf(mx, w[f]); }
  float sum = 0.f;
  for (int f = 0; f < kF; ++f) { w[f] = __expf(w[f] - mx); sum += w[f]; }
  const float inv = 1.f / sum;

  const float smod = 1.f + style[b * kD + i];
  float vals[kTaps];
  float ss = 0.f;
  for (int t = 0; t < kTaps; ++t) {
    float acc = 0.f;
    for (int f = 0; f < kF; ++f)
      acc += (w[f] * inv) * bank_weight[(((size_t)f * kD + o) * kD + i) * kTaps + t];
    const float v = acc * smod;
    vals[t] = v;
    ss += v * v;
  }

  __shared__ float red[256];
  red[i] = ss;
  __syncthreads();
  for (int s = 128; s > 0; s >>= 1) {
    if (i < s) red[i] += red[i + s];
    __syncthreads();
  }
  const float nrm = rsqrtf(red[0] + 1e-8f);

  for (int t = 0; t < kTaps; ++t)
    Kw[(((size_t)(b * kD + o)) * kTaps + t) * kD + i] = (__bf16)(vals[t] * nrm);
}

// ---------------------------------------------------------------------------
// Kernel 2: x (f32, NCHW) -> bf16, same layout.
// ---------------------------------------------------------------------------
__global__ __launch_bounds__(256) void convert_x(
    const float* __restrict__ x, __bf16* __restrict__ xb)
{
  const int idx = blockIdx.x * 256 + threadIdx.x;
  const float4 v = ((const float4*)x)[idx];
  v4bf o;
  o[0] = (__bf16)v.x; o[1] = (__bf16)v.y; o[2] = (__bf16)v.z; o[3] = (__bf16)v.w;
  ((v4bf*)xb)[idx] = o;
}

// ---------------------------------------------------------------------------
// Kernel 3: implicit-GEMM conv via v_wmma_f32_16x16x32_bf16.
// Block tile: 128 output channels x one full image row (N = 64 pixels).
// 8 waves: wave w owns M tile o0 + w*16, computes 4 16x16 accumulators
// spanning the whole row, so each A fragment feeds 4 WMMAs.
// Fully unrolled 3x3 tap loops: 36 static WMMAs per K chunk, all independent
// accumulator chains (no WMMA->WMMA D->A/B hazards).
// ---------------------------------------------------------------------------
__global__ __launch_bounds__(256) void conv_wmma(
    const __bf16* __restrict__ Xbf,
    const __bf16* __restrict__ Kw,
    float* __restrict__ out)
{
  __shared__ __bf16 Xs[3 * XS * LPAD];  // [yrow 0..2][x-slot 0..65][i 0..39]

  const int tid  = threadIdx.x;
  const int lane = tid & 31;
  const int wave = tid >> 5;          // 0..7 : 16-row M tile
  const int y    = blockIdx.x;        // image row
  const int o0   = blockIdx.y * 128;  // output-channel tile base
  const int b    = blockIdx.z;

  const int mrow  = lane & 15;
  const int kselA = (lane < 16) ? 0 : 8;   // 16-bit A 16x32 lane layout
  const int kselB = (lane < 16) ? 0 : 8;   // 16-bit B 32x16 lane layout

  const __bf16* xbase = Xbf + (size_t)b * kD * kH * kW;
  const __bf16* krow  = Kw + ((size_t)(b * kD + o0 + wave * 16 + mrow)) * kKlen;

  v8f acc[4] = {{}, {}, {}, {}};

  // Halo columns (x = -1 and x = 64) are zero for every channel chunk:
  // write them once. e = side + 2*(i + 32*r), all shift/mask decompose.
  for (int e = tid; e < 3 * 32 * 2; e += 256) {
    const int side = e & 1;
    const int ri   = e >> 1;
    const int i    = ri & 31;
    const int r    = ri >> 5;
    Xs[(r * XS + side * (XS - 1)) * LPAD + i] = (__bf16)0.f;
  }

  for (int c = 0; c < kD / 32; ++c) {
    const int i0 = c * 32;
    __syncthreads();
    // Stage interior: 3 rows x 32 channels x 8 chunks of 8 pixels (16B each).
    // e = g + 8*(i + 32*r): consecutive tids load consecutive 16B -> coalesced.
    for (int e = tid; e < 3 * 32 * 8; e += 256) {
      const int g  = e & 7;        // 8-pixel group: x = g*8 .. g*8+7
      const int ri = e >> 3;
      const int i  = ri & 31;
      const int r  = ri >> 5;
      const int yy = y + r - 1;
      const bool inb = (yy >= 0) && (yy < kH);
      if (inb || c == 0) {
        v8bf v = {};
        if (inb)
          v = *(const v8bf*)(xbase + ((size_t)(i0 + i) * kH + yy) * kW + g * 8);
        #pragma unroll
        for (int q = 0; q < 8; ++q)
          Xs[(r * XS + g * 8 + q + 1) * LPAD + i] = v[q];
      }
    }
    if (c + 1 < kD / 32)  // pull next channel chunk toward L2 (global_prefetch_b8)
      __builtin_prefetch(xbase + ((size_t)(i0 + 32 + lane) * kH + y) * kW, 0, 1);
    __syncthreads();

    // Per-lane invariant LDS/global bases for this chunk.
    const __bf16* kbase   = krow + i0 + kselA;
    const int     sfirst  = (lane & 15);  // N-subtile column within slot row

    #pragma unroll
    for (int dy = 0; dy < 3; ++dy) {
      #pragma unroll
      for (int dx = 0; dx < 3; ++dx) {
        const int t = dy * 3 + dx;

        // A fragment: 16(o) x 32(i) kernel tile for tap t, contiguous in Kw.
        const __bf16* aptr = kbase + t * kD;            // constant offset fold
        const v8bf alo = *(const v8bf*)aptr;
        const v8bf ahi = *(const v8bf*)(aptr + 16);
        v16bf a;
        #pragma unroll
        for (int q = 0; q < 8; ++q) { a[q] = alo[q]; a[q + 8] = ahi[q]; }

        // 4 B fragments: 32(i) x 16(x), contiguous 16B LDS reads per lane.
        #pragma unroll
        for (int s = 0; s < 4; ++s) {
          const int scol = s * 16 + sfirst + dx;        // x-slot = x + dx
          const __bf16* p = &Xs[(dy * XS + scol) * LPAD + kselB];
          const v8bf blo = *(const v8bf*)p;
          const v8bf bhi = *(const v8bf*)(p + 16);
          v16bf bv;
          #pragma unroll
          for (int q = 0; q < 8; ++q) { bv[q] = blo[q]; bv[q + 8] = bhi[q]; }
          acc[s] = __builtin_amdgcn_wmma_f32_16x16x32_bf16(
              false, a, false, bv, (short)0, acc[s], false, false);
        }
      }
    }
  }

  // C/D 16x16 f32 layout: VGPR r -> row r (+8 for lanes 16..31), col = lane%16.
  const int n    = lane & 15;
  const int moff = (lane < 16) ? 0 : 8;
  float* obase = out + (((size_t)(b * kD + o0 + wave * 16 + moff)) * kH + y) * kW + n;
  #pragma unroll
  for (int s = 0; s < 4; ++s)
    #pragma unroll
    for (int r = 0; r < 8; ++r)
      obase[(size_t)r * kH * kW + s * 16] = acc[s][r];
}

// ---------------------------------------------------------------------------
extern "C" void kernel_launch(void* const* d_in, const int* in_sizes, int n_in,
                              void* d_out, int out_size, void* d_ws, size_t ws_size,
                              hipStream_t stream) {
  const float* x            = (const float*)d_in[0];
  const float* bank_request = (const float*)d_in[1];
  const float* style        = (const float*)d_in[2];
  const float* bank_weight  = (const float*)d_in[3];
  float* out = (float*)d_out;

  const size_t kwBytes = (size_t)kB * kD * kKlen * sizeof(__bf16);  // 9,437,184
  __bf16* Kw  = (__bf16*)d_ws;
  __bf16* Xbf = (__bf16*)((char*)d_ws + kwBytes);                   // +16,777,216

  build_kernels<<<dim3(kD, kB), 256, 0, stream>>>(bank_request, style, bank_weight, Kw);

  const int nX = kB * kD * kH * kW;             // 8,388,608 floats
  convert_x<<<nX / (256 * 4), 256, 0, stream>>>(x, Xbf);

  conv_wmma<<<dim3(kH, kD / 128, kB), 256, 0, stream>>>(Xbf, Kw, out);
}
